// SlabAttention_89850715833027
// MI455X (gfx1250) — matrix operand order repaired and checked by reading, not compiled, not run on hardware
//
#include <hip/hip_runtime.h>
#include <hip/hip_bf16.h>

typedef __attribute__((ext_vector_type(16))) _Float16 v16h;
typedef __attribute__((ext_vector_type(8)))  _Float16 v8h;
typedef __attribute__((ext_vector_type(8)))  float    v8f;
typedef __attribute__((ext_vector_type(4)))  unsigned int v4u;
typedef __attribute__((ext_vector_type(8)))  int v8i;
typedef __attribute__((ext_vector_type(4)))  int v4i;

#define NHEADS 12
#define HD     32
#define BATCH  16
#define NTOK   4096
#define CDIM   384
#define BHDIM  (BATCH * NHEADS)   // 192
#define EPSV   1e-6f

// LDS A-tile: 128 rows x 32 halves, padded to 40 halves (80B) to dodge bank conflicts
#define APAD 40

// ---------- CDNA5 async-copy helpers ----------
__device__ __forceinline__ void async_load_b128(uint32_t lds_off, const void* gptr) {
  asm volatile("global_load_async_to_lds_b128 %0, %1, off"
               :: "v"(lds_off), "v"((unsigned long long)(uintptr_t)gptr)
               : "memory");
}
__device__ __forceinline__ void wait_async0() {
  asm volatile("s_wait_asynccnt 0" ::: "memory");
}

// ---------- CDNA5 Tensor Data Mover: 2D tile (rows x 32 halves) -> padded LDS ----------
// D# per CDNA5 ISA. Pads 4 DWORDs every 16 DWORDs (64B row)
// -> LDS row stride 80B, matching As[...][APAD=40].
__device__ __forceinline__ void tdm_load_tile_a32(uint32_t lds_base, const _Float16* gbase) {
  const unsigned long long ga = (unsigned long long)(uintptr_t)gbase;
  v4u g0;
  g0[0] = 1u;                                              // count=1, user descriptor
  g0[1] = lds_base;                                        // lds_addr [63:32]
  g0[2] = (unsigned)(ga & 0xFFFFFFFFu);                    // global_addr[31:0]
  g0[3] = (unsigned)((ga >> 32) & 0x1FFFFFFu) | (2u << 30);// global_addr[56:32] | type=2
  v8i g1;
  g1[0] = (1 << 16)      // data_size = 1 (2 bytes)
        | (1 << 20)      // pad_enable
        | (3 << 22)      // pad_interval = 16 DWORDs (one 64B row)
        | (3 << 25);     // pad_amount   = 4 DWORDs (16B)
  g1[1] = (int)(32u << 16);   // tensor_dim0 = 32  ([79:48], low half here)
  g1[2] = (int)(128u << 16);  // tensor_dim1 = 128 ([111:80], low half here)
  g1[3] = (int)(32u << 16);   // tile_dim0 = 32    ([127:112])
  g1[4] = 128;                // tile_dim1 = 128   ([143:128]); tile_dim2 = 0
  g1[5] = 384;                // tensor_dim0_stride = 384 ([207:160], low 32)
  g1[6] = 0;
  g1[7] = 0;
  const v4i gz = {0, 0, 0, 0};
#if __clang_major__ >= 23
  const v8i gz8 = {0, 0, 0, 0, 0, 0, 0, 0};
  __builtin_amdgcn_tensor_load_to_lds(g0, g1, gz, gz, gz8, 0);
#else
  __builtin_amdgcn_tensor_load_to_lds(g0, g1, gz, gz, 0);
#endif
}

// A-fragment for v_wmma_f32_16x16x32_f16 from a row pointer with K contiguous:
// lane hi = lane>>4; a[0..7] = A[m][hi*16 + j]; a[8..15] = A[m][hi*16 + 8 + j]
__device__ __forceinline__ v16h make_a16(const _Float16* rowp, int hi) {
  const v8h lo = *(const v8h*)(rowp + hi * 16);
  const v8h hh = *(const v8h*)(rowp + hi * 16 + 8);
  v16h a;
#pragma unroll
  for (int i = 0; i < 8; ++i) { a[i] = lo[i]; a[i + 8] = hh[i]; }
  return a;
}

__device__ __forceinline__ v8f wmma_f16(v16h a, v16h b, v8f c) {
  return __builtin_amdgcn_wmma_f32_16x16x32_f16(false, a, false, b, (short)0, c, false, false);
}

// ---------- kernel 0: f32 -> f16 convert ----------
__global__ __launch_bounds__(256) void cvt_f16(const float* __restrict__ src,
                                               _Float16* __restrict__ dst, int n) {
  int i = blockIdx.x * 256 + threadIdx.x;
  if (i < n) dst[i] = (_Float16)src[i];
}

// ---------- kernel 0b: zero f32 buffer ----------
__global__ __launch_bounds__(256) void zero_f32(float* __restrict__ p, int n) {
  int i = blockIdx.x * 256 + threadIdx.x;
  if (i < n) p[i] = 0.f;
}

// ---------- kernel 1: qkv GEMM (async-LDS staged A, direct B) + fused epilogue ----------
__global__ __launch_bounds__(256) void qkv_gemm(const _Float16* __restrict__ xh,
                                                const _Float16* __restrict__ wh,
                                                const float* __restrict__ pos,
                                                _Float16* __restrict__ qh,
                                                _Float16* __restrict__ kh,
                                                _Float16* __restrict__ vh) {
  __shared__ __align__(64) _Float16 As[2][128][APAD];
  const int lane = threadIdx.x & 31, wave = threadIdx.x >> 5;
  const int mwave = wave & 3, nwave = wave >> 2;          // 4 waves M x 2 waves N
  const int mblk = blockIdx.y * 128;
  const int m0 = mwave * 32;                              // within block
  const int n0 = blockIdx.x * 128 + nwave * 64;
  const int l16 = lane & 15, hi = lane >> 4;

  // per-thread staging assignment: 2 x b128 chunks covering 128x32 halves
  const int srow = threadIdx.x >> 1;                      // 0..127
  const int sc0  = (threadIdx.x & 1) ? 1 : 0;             // chunks {0,2} or {1,3}

  // prologue stage for k0 = 0
#pragma unroll
  for (int ci = 0; ci < 2; ++ci) {
    const int c4 = sc0 + ci * 2;
    async_load_b128((uint32_t)(uintptr_t)&As[0][srow][c4 * 8],
                    xh + (size_t)(mblk + srow) * CDIM + 0 + c4 * 8);
  }
  wait_async0();
  __syncthreads();

  v8f acc[2][4] = {};
#pragma unroll 1
  for (int k0 = 0; k0 < CDIM; k0 += 32) {
    const int buf = (k0 >> 5) & 1;
    const bool more = (k0 + 32 < CDIM);
    if (more) {
#pragma unroll
      for (int ci = 0; ci < 2; ++ci) {
        const int c4 = sc0 + ci * 2;
        async_load_b128((uint32_t)(uintptr_t)&As[buf ^ 1][srow][c4 * 8],
                        xh + (size_t)(mblk + srow) * CDIM + (k0 + 32) + c4 * 8);
      }
      __builtin_prefetch(wh + (size_t)(n0 + l16) * CDIM + k0 + 32, 0, 1);
    }
    v16h a[2], b[4];
#pragma unroll
    for (int mt = 0; mt < 2; ++mt)
      a[mt] = make_a16(&As[buf][m0 + mt * 16 + l16][0], hi);
#pragma unroll
    for (int nt = 0; nt < 4; ++nt)
      b[nt] = *(const v16h*)(wh + (size_t)(n0 + nt * 16 + l16) * CDIM + k0 + hi * 16);
#pragma unroll
    for (int mt = 0; mt < 2; ++mt)
#pragma unroll
      for (int nt = 0; nt < 4; ++nt)
        acc[mt][nt] = wmma_f16(a[mt], b[nt], acc[mt][nt]);
    if (more) {
      wait_async0();
      __syncthreads();
    }
  }

  // epilogue: split q/k/v, relu, +pos_enc, head-major store
#pragma unroll
  for (int nt = 0; nt < 4; ++nt) {
    const int n = n0 + nt * 16 + l16;  // 0..1151
#pragma unroll
    for (int mt = 0; mt < 2; ++mt) {
#pragma unroll
      for (int r = 0; r < 8; ++r) {
        const int row  = mblk + m0 + mt * 16 + r + 8 * hi;
        const int bidx = row >> 12;        // /4096
        const int nrow = row & 4095;
        float val = acc[mt][nt][r];
        if (n < CDIM) {
          const int h = n >> 5, d = n & 31;
          val = val > 0.f ? val : 0.f;
          qh[(((size_t)bidx * NHEADS + h) * NTOK + nrow) * HD + d] = (_Float16)val;
        } else if (n < 2 * CDIM) {
          const int cc = n - CDIM, h = cc >> 5, d = cc & 31;
          val += pos[(size_t)nrow * CDIM + cc];
          val = val > 0.f ? val : 0.f;
          kh[(((size_t)bidx * NHEADS + h) * NTOK + nrow) * HD + d] = (_Float16)val;
        } else {
          const int cc = n - 2 * CDIM, h = cc >> 5, d = cc & 31;
          vh[(((size_t)bidx * NHEADS + h) * NTOK + nrow) * HD + d] = (_Float16)val;
        }
      }
    }
  }
}

// ---------- kernel 2: ksum[bh][c] = sum_j k[bh][j][c] ----------
__global__ __launch_bounds__(256) void ksum_kernel(const _Float16* __restrict__ kh,
                                                   float* __restrict__ ksum) {
  const int bh = blockIdx.x, t = threadIdx.x;
  const int c = t & 31, jg = t >> 5;
  const _Float16* kp = kh + (size_t)bh * NTOK * HD;
  float s = 0.f;
  for (int j = jg; j < NTOK; j += 8) s += (float)kp[(size_t)j * HD + c];
  __shared__ float sm[256];
  sm[t] = s;
  __syncthreads();
  if (t < 32) {
    float tot = 0.f;
#pragma unroll
    for (int g = 0; g < 8; ++g) tot += sm[g * 32 + t];
    ksum[bh * HD + t] = tot;
  }
}

// ---------- kernel 3: z[bh][i] = 1/(q_i . ksum + eps) ----------
__global__ __launch_bounds__(256) void z_kernel(const _Float16* __restrict__ qh,
                                                const float* __restrict__ ksum,
                                                float* __restrict__ z) {
  const size_t idx = (size_t)blockIdx.x * 256 + threadIdx.x;  // BH*NTOK
  const int bh = (int)(idx >> 12), i = (int)(idx & 4095);
  const _Float16* qp = qh + ((size_t)bh * NTOK + i) * HD;
  const float* ks = ksum + bh * HD;
  float s = 0.f;
#pragma unroll
  for (int c = 0; c < HD; ++c) s += (float)qp[c] * ks[c];
  z[idx] = 1.f / (s + EPSV);
}

// ---------- kernel 4a: kv partials, j split 8-way, atomic f32 accumulate ----------
// kvacc[bh][d][c] += sum_j k[j][c] * v[j][d]
__global__ __launch_bounds__(256) void kv_partial(const _Float16* __restrict__ kh,
                                                  const _Float16* __restrict__ vh,
                                                  float* __restrict__ kvacc) {
  const int bh = blockIdx.x, jc = blockIdx.y;
  const int t = threadIdx.x;
  const int c = t & 31, dg = t >> 5;  // thread owns (c, d = dg*4 .. dg*4+3)
  const _Float16* kp = kh + ((size_t)bh * NTOK + jc * 512) * HD;
  const _Float16* vp = vh + ((size_t)bh * NTOK + jc * 512) * HD;
  float acc[4] = {0.f, 0.f, 0.f, 0.f};
  for (int j = 0; j < 512; ++j) {
    const float kv_ = (float)kp[(size_t)j * HD + c];
#pragma unroll
    for (int u = 0; u < 4; ++u) acc[u] += kv_ * (float)vp[(size_t)j * HD + dg * 4 + u];
  }
#pragma unroll
  for (int u = 0; u < 4; ++u)
    atomicAdd(&kvacc[(size_t)bh * HD * HD + (dg * 4 + u) * HD + c], acc[u]);
}

// ---------- kernel 4b: kvT (f16) = kvacc (f32), same [bh][d][c] layout ----------
__global__ __launch_bounds__(256) void kv_cvt(const float* __restrict__ kvacc,
                                              _Float16* __restrict__ kvT, int n) {
  int i = blockIdx.x * 256 + threadIdx.x;
  if (i < n) kvT[i] = (_Float16)kvacc[i];
}

// ---------- kernel 5: depthwise 5x5 conv on v -> att (f16, (B,N,C) layout) ----------
__global__ __launch_bounds__(256) void dwconv_kernel(const _Float16* __restrict__ vh,
                                                     const float* __restrict__ dwc_w,
                                                     const float* __restrict__ dwc_b,
                                                     _Float16* __restrict__ att) {
  const size_t idx = (size_t)blockIdx.x * 256 + threadIdx.x;  // BH*NTOK*HD
  const int d = (int)(idx & 31);
  const size_t r = idx >> 5;
  const int n = (int)(r & 4095);
  const int bh = (int)(r >> 12);
  const int i = n >> 6, j = n & 63;
  const _Float16* vp = vh + (size_t)bh * NTOK * HD;
  float s = dwc_b[d];
#pragma unroll
  for (int di = 0; di < 5; ++di) {
    const int ii = i + di - 2;
    if (ii < 0 || ii >= 64) continue;
#pragma unroll
    for (int dj = 0; dj < 5; ++dj) {
      const int jj = j + dj - 2;
      if (jj < 0 || jj >= 64) continue;
      s += dwc_w[d * 25 + di * 5 + dj] * (float)vp[((size_t)ii * 64 + jj) * HD + d];
    }
  }
  const int b = bh / NHEADS, h = bh % NHEADS;
  att[((size_t)(b * NTOK + n)) * CDIM + h * HD + d] = (_Float16)s;
}

// ---------- kernel 6: att += (q @ kv) * z (WMMA, K=32 single step) ----------
__global__ __launch_bounds__(256) void attn_out_kernel(const _Float16* __restrict__ qh,
                                                       const _Float16* __restrict__ kvT,
                                                       const float* __restrict__ z,
                                                       _Float16* __restrict__ att) {
  const int bh = blockIdx.x, rb = blockIdx.y;
  const int lane = threadIdx.x & 31, wave = threadIdx.x >> 5;
  const int row0 = rb * 512 + wave * 64;  // wave tile: 64 rows x 32 cols
  const int l16 = lane & 15, hi = lane >> 4;
  const _Float16* qp = qh + (size_t)bh * NTOK * HD;
  const _Float16* kp = kvT + (size_t)bh * HD * HD;

  v16h b[2];
#pragma unroll
  for (int nt = 0; nt < 2; ++nt)
    b[nt] = *(const v16h*)(kp + (size_t)(nt * 16 + l16) * HD + hi * 16);

  v16h a[4];
#pragma unroll
  for (int mt = 0; mt < 4; ++mt)
    a[mt] = make_a16(qp + (size_t)(row0 + mt * 16 + l16) * HD, hi);

  v8f acc[4][2] = {};
#pragma unroll
  for (int mt = 0; mt < 4; ++mt)
#pragma unroll
    for (int nt = 0; nt < 2; ++nt)
      acc[mt][nt] = wmma_f16(a[mt], b[nt], acc[mt][nt]);

  const int bb = bh / NHEADS, h = bh % NHEADS;
  const float* zp = z + (size_t)bh * NTOK;
#pragma unroll
  for (int mt = 0; mt < 4; ++mt) {
#pragma unroll
    for (int nt = 0; nt < 2; ++nt) {
      const int d = nt * 16 + l16;
#pragma unroll
      for (int r = 0; r < 8; ++r) {
        const int row = row0 + mt * 16 + r + 8 * hi;
        const float val = acc[mt][nt][r] * zp[row];
        const size_t o = ((size_t)(bb * NTOK + row)) * CDIM + h * HD + d;
        att[o] = (_Float16)((float)att[o] + val);
      }
    }
  }
}

// ---------- kernel 7: out = att @ proj_w^T + proj_b (TDM-staged A tile) ----------
__global__ __launch_bounds__(256) void proj_gemm(const _Float16* __restrict__ att,
                                                 const _Float16* __restrict__ pwh,
                                                 const float* __restrict__ pb,
                                                 float* __restrict__ out) {
  __shared__ __align__(64) _Float16 As[2][128][APAD];
  const int lane = threadIdx.x & 31, wave = threadIdx.x >> 5;
  const int mwave = wave & 3, nwave = wave >> 2;
  const int mblk = blockIdx.y * 128;
  const int m0 = mwave * 32;
  const int n0 = blockIdx.x * 128 + nwave * 64;
  const int l16 = lane & 15, hi = lane >> 4;

  // prologue: TDM DMA of A tile (128 rows x 32 halves) for k0 = 0, wave 0 only
  if (wave == 0) {
    tdm_load_tile_a32((uint32_t)(uintptr_t)&As[0][0][0],
                      att + (size_t)mblk * CDIM + 0);
    __builtin_amdgcn_s_wait_tensorcnt(0);
  }
  __syncthreads();

  v8f acc[2][4] = {};
#pragma unroll 1
  for (int k0 = 0; k0 < CDIM; k0 += 32) {
    const int buf = (k0 >> 5) & 1;
    const bool more = (k0 + 32 < CDIM);
    if (more) {
      if (wave == 0)
        tdm_load_tile_a32((uint32_t)(uintptr_t)&As[buf ^ 1][0][0],
                          att + (size_t)mblk * CDIM + (k0 + 32));
      __builtin_prefetch(pwh + (size_t)(n0 + l16) * CDIM + k0 + 32, 0, 1);
    }
    v16h a[2], b[4];
#pragma unroll
    for (int mt = 0; mt < 2; ++mt)
      a[mt] = make_a16(&As[buf][m0 + mt * 16 + l16][0], hi);
#pragma unroll
    for (int nt = 0; nt < 4; ++nt)
      b[nt] = *(const v16h*)(pwh + (size_t)(n0 + nt * 16 + l16) * CDIM + k0 + hi * 16);
#pragma unroll
    for (int mt = 0; mt < 2; ++mt)
#pragma unroll
      for (int nt = 0; nt < 4; ++nt)
        acc[mt][nt] = wmma_f16(a[mt], b[nt], acc[mt][nt]);
    if (more) {
      if (wave == 0) __builtin_amdgcn_s_wait_tensorcnt(0);
      __syncthreads();
    }
  }

#pragma unroll
  for (int nt = 0; nt < 4; ++nt) {
    const int n = n0 + nt * 16 + l16;
    const float bias = pb[n];
#pragma unroll
    for (int mt = 0; mt < 2; ++mt) {
#pragma unroll
      for (int r = 0; r < 8; ++r) {
        const int row = mblk + m0 + mt * 16 + r + 8 * hi;
        out[(size_t)row * CDIM + n] = acc[mt][nt][r] + bias;
      }
    }
  }
}

// ---------- host ----------
extern "C" void kernel_launch(void* const* d_in, const int* in_sizes, int n_in,
                              void* d_out, int out_size, void* d_ws, size_t ws_size,
                              hipStream_t stream) {
  (void)in_sizes; (void)n_in; (void)out_size; (void)ws_size;
  const float* x      = (const float*)d_in[0];
  const float* qkv_w  = (const float*)d_in[1];
  const float* pos    = (const float*)d_in[2];
  const float* dwc_w  = (const float*)d_in[3];
  const float* dwc_b  = (const float*)d_in[4];
  const float* proj_w = (const float*)d_in[5];
  const float* proj_b = (const float*)d_in[6];
  float* out = (float*)d_out;

  char* ws = (char*)d_ws;
  const size_t M  = (size_t)BATCH * NTOK;          // 65536
  const size_t XE = M * CDIM;                      // 25,165,824 elems
  const size_t HE = (size_t)BHDIM * NTOK * HD;     // 25,165,824 elems

  size_t off = 0;
  _Float16* xh     = (_Float16*)(ws + off);                    // aliased by att
  _Float16* att    = (_Float16*)(ws + off);  off += 2 * XE;    // xh dead after qkv_gemm
  _Float16* qh     = (_Float16*)(ws + off);  off += 2 * XE;
  _Float16* kh     = (_Float16*)(ws + off);  off += 2 * XE;
  _Float16* vh     = (_Float16*)(ws + off);  off += 2 * XE;
  _Float16* qkvwh  = (_Float16*)(ws + off);  off += 1024 * 1024;  // 442368 elems
  _Float16* projwh = (_Float16*)(ws + off);  off += 512 * 1024;   // 147456 elems
  float*    ksum   = (float*)   (ws + off);  off += 32 * 1024;    // 192*32
  float*    zbuf   = (float*)   (ws + off);  off += 4 * (size_t)BHDIM * NTOK; // 192*4096 f32
  _Float16* kvT    = (_Float16*)(ws + off);  off += 2 * (size_t)BHDIM * HD * HD;
  float*    kvacc  = (float*)   (ws + off);  off += 4 * (size_t)BHDIM * HD * HD;

  const int KVN = BHDIM * HD * HD;  // 196608

  // 1) f32 -> f16 conversions
  cvt_f16<<<(int)((XE + 255) / 256), 256, 0, stream>>>(x, xh, (int)XE);
  cvt_f16<<<(3 * CDIM * CDIM + 255) / 256, 256, 0, stream>>>(qkv_w, qkvwh, 3 * CDIM * CDIM);
  cvt_f16<<<(CDIM * CDIM + 255) / 256, 256, 0, stream>>>(proj_w, projwh, CDIM * CDIM);

  // 2) qkv GEMM + epilogue   (grid: 1152/128 x 65536/128)
  qkv_gemm<<<dim3(9, 512), 256, 0, stream>>>(xh, qkvwh, pos, qh, kh, vh);

  // 3) ksum, z
  ksum_kernel<<<BHDIM, 256, 0, stream>>>(kh, ksum);
  z_kernel<<<(int)((size_t)BHDIM * NTOK / 256), 256, 0, stream>>>(qh, ksum, zbuf);

  // 4) kv = k^T v  (8-way j-split partials + f32 atomics, then f16 convert)
  zero_f32<<<(KVN + 255) / 256, 256, 0, stream>>>(kvacc, KVN);
  kv_partial<<<dim3(BHDIM, 8), 256, 0, stream>>>(kh, vh, kvacc);
  kv_cvt<<<(KVN + 255) / 256, 256, 0, stream>>>(kvacc, kvT, KVN);

  // 5) depthwise conv -> att (writes every element; overwrites xh alias)
  dwconv_kernel<<<(int)((HE + 255) / 256), 256, 0, stream>>>(vh, dwc_w, dwc_b, att);

  // 6) att += (q @ kv) * z
  attn_out_kernel<<<dim3(BHDIM, 8), 256, 0, stream>>>(qh, kvT, zbuf, att);

  // 7) final projection -> fp32 out   (grid: 384/128 x 65536/128)
  proj_gemm<<<dim3(3, 512), 256, 0, stream>>>(att, projwh, proj_b, out);
}
